// ResnetCnnsovnetDynamicRouting_30185030156913
// MI455X (gfx1250) — compile-verified, matrix operand order checked by reading, not run on hardware
//
#include <hip/hip_runtime.h>
#include <math.h>
#include <stdint.h>

typedef __attribute__((ext_vector_type(16))) _Float16 v16h;
typedef __attribute__((ext_vector_type(8)))  _Float16 v8h;
typedef __attribute__((ext_vector_type(8)))  float    v8f;
typedef __attribute__((ext_vector_type(4)))  int      v4i;

#if defined(__has_builtin)
#if __has_builtin(__builtin_amdgcn_global_load_async_to_lds_b128)
#define ASYNC_LDS 1
#endif
#endif

#define ASG __attribute__((address_space(1)))
#define ASL __attribute__((address_space(3)))

// ---------------------------------------------------------------------------
// Implicit-GEMM convolution with fused BN/residual/ReLU epilogue.
// Block: 256 threads = 8 wave32s, arranged 2(M) x 4(N).
// Block tile 64(Cout) x 128(pixels); each wave computes 32x32 via four
// v_wmma_f32_16x16x32_f16 per K-step. Weights are pre-padded to Kpad=ceil32(K)
// so the A tile streams with unconditional 16B copies (async-to-LDS on
// toolchains that expose the gfx1250 builtin).
// All Ho*Wo / Wo are powers of two in this network -> shifts; khw/kw use
// exact 16-bit magic reciprocals.
// ---------------------------------------------------------------------------
struct ConvP {
  const _Float16* x;     // (N, Cin, H, W)
  const _Float16* w;     // (Cout, Kpad) padded
  const float*    scale;
  const float*    bias;
  const _Float16* res;   // optional residual (N, Cout, Ho, Wo)
  _Float16*       y;     // (N, Cout, Ho, Wo)
  int N, Cin, H, W, Cout, K, Kpad, kh, kw, stride, pad, Ho, Wo, relu;
  int howo_sh, wo_sh;          // log2(Ho*Wo), log2(Wo)
  unsigned kinv_khw, kinv_kw;  // ceil(2^16 / d) exact reciprocals
};

__global__ __launch_bounds__(256) void conv_wmma(ConvP p) {
  __shared__ _Float16 As[64 * 32];    // [m][k]
  __shared__ _Float16 Bs[128 * 32];   // [n][k] (transposed store)
  const int tid  = threadIdx.x;
  const int lane = tid & 31;
  const int wv   = tid >> 5;
  const int wn   = wv & 3;            // N-tile 0..3 (x32)
  const int wm   = wv >> 2;           // M-tile 0..1 (x32)
  const int HoWo = p.Ho * p.Wo;
  const int Ntot = p.N * HoWo;
  const int n0 = blockIdx.x * 128;
  const int m0 = blockIdx.y * 64;
  const int khw = p.kh * p.kw;

  // ---- B-tile loader mapping: each thread fills 16 k for one n column ----
  const int bn_ = tid & 127;
  const int bks = (tid >> 7) << 4;    // 0 or 16
  const int ng  = n0 + bn_;
  int batch = 0, ihb = 0, iwb = 0;
  const bool nvalid = ng < Ntot;
  if (nvalid) {
    batch = ng >> p.howo_sh;
    int r = ng & (HoWo - 1);
    int oh = r >> p.wo_sh, ow = r & (p.Wo - 1);
    ihb = oh * p.stride - p.pad;
    iwb = ow * p.stride - p.pad;
  }
  // ---- A-tile loader mapping: each thread copies 16B (8 f16) ----
  const int am = tid >> 2;
  const int ak = (tid & 3) << 3;
  const int cg = m0 + am;
  const int cgc = cg < p.Cout ? cg : (p.Cout - 1);   // clamp: always in-bounds

  v8f acc[4];
#pragma unroll
  for (int q = 0; q < 4; ++q) acc[q] = (v8f){0.f, 0.f, 0.f, 0.f, 0.f, 0.f, 0.f, 0.f};

  for (int k0 = 0; k0 < p.Kpad; k0 += 32) {
    // ---- stage A (weights, padded: unconditional 16B transfer) ----
    const _Float16* gsrc = p.w + (size_t)cgc * p.Kpad + k0 + ak;
    _Float16* ldst = &As[am * 32 + ak];
#ifdef ASYNC_LDS
    __builtin_amdgcn_global_load_async_to_lds_b128(
        (ASG v4i*)(uintptr_t)gsrc,
        (ASL v4i*)(unsigned)(uintptr_t)ldst, 0, 0);
#else
    *(v8h*)ldst = *(const v8h*)gsrc;
#endif
    if (k0 + 32 < p.Kpad)
      __builtin_prefetch((const void*)(gsrc + 32), 0, 1);

    // ---- stage B (im2col gather, zero padded) ----
    v8h blo, bhi;
#pragma unroll
    for (int u = 0; u < 8; ++u) {
      int k  = k0 + bks + u;
      int k2 = k + 8;
      _Float16 v0 = (_Float16)0.f, v1 = (_Float16)0.f;
      if (nvalid && k < p.K) {
        int ci = (int)(((unsigned)k * p.kinv_khw) >> 16);
        int rr = k - ci * khw;
        int rq = (int)(((unsigned)rr * p.kinv_kw) >> 16);
        int sx = rr - rq * p.kw;
        int ih = ihb + rq, iw = iwb + sx;
        if (ih >= 0 && ih < p.H && iw >= 0 && iw < p.W)
          v0 = p.x[((size_t)(batch * p.Cin + ci) * p.H + ih) * p.W + iw];
      }
      if (nvalid && k2 < p.K) {
        int ci = (int)(((unsigned)k2 * p.kinv_khw) >> 16);
        int rr = k2 - ci * khw;
        int rq = (int)(((unsigned)rr * p.kinv_kw) >> 16);
        int sx = rr - rq * p.kw;
        int ih = ihb + rq, iw = iwb + sx;
        if (ih >= 0 && ih < p.H && iw >= 0 && iw < p.W)
          v1 = p.x[((size_t)(batch * p.Cin + ci) * p.H + ih) * p.W + iw];
      }
      blo[u] = v0;
      bhi[u] = v1;
    }
    *(v8h*)&Bs[bn_ * 32 + bks]     = blo;
    *(v8h*)&Bs[bn_ * 32 + bks + 8] = bhi;

#ifdef ASYNC_LDS
#if __has_builtin(__builtin_amdgcn_s_wait_asynccnt)
    __builtin_amdgcn_s_wait_asynccnt(0);
#else
    asm volatile("s_wait_asynccnt 0x0" ::: "memory");
#endif
#endif
    __syncthreads();

    // ---- fragments per CDNA5 16-bit 16x32 layout ----
    // lanes 0-15: K kb..kb+7 -> a[0..7], K kb+16..kb+23 -> a[8..15]; lanes 16-31: kb=8.
    const int arow = (wm << 5) + (lane & 15);
    const int kb   = (lane >> 4) << 3;
    v8h al0 = *(const v8h*)&As[arow * 32 + kb];
    v8h ah0 = *(const v8h*)&As[arow * 32 + kb + 16];
    v8h al1 = *(const v8h*)&As[(arow + 16) * 32 + kb];
    v8h ah1 = *(const v8h*)&As[(arow + 16) * 32 + kb + 16];
    v16h a0 = __builtin_shufflevector(al0, ah0, 0,1,2,3,4,5,6,7,8,9,10,11,12,13,14,15);
    v16h a1 = __builtin_shufflevector(al1, ah1, 0,1,2,3,4,5,6,7,8,9,10,11,12,13,14,15);
    const int bc = (wn << 5) + (lane & 15);
    v8h bl0 = *(const v8h*)&Bs[bc * 32 + kb];
    v8h bh0 = *(const v8h*)&Bs[bc * 32 + kb + 16];
    v8h bl1 = *(const v8h*)&Bs[(bc + 16) * 32 + kb];
    v8h bh1 = *(const v8h*)&Bs[(bc + 16) * 32 + kb + 16];
    v16h b0 = __builtin_shufflevector(bl0, bh0, 0,1,2,3,4,5,6,7,8,9,10,11,12,13,14,15);
    v16h b1 = __builtin_shufflevector(bl1, bh1, 0,1,2,3,4,5,6,7,8,9,10,11,12,13,14,15);

    acc[0] = __builtin_amdgcn_wmma_f32_16x16x32_f16(false, a0, false, b0, (short)0, acc[0], false, false);
    acc[1] = __builtin_amdgcn_wmma_f32_16x16x32_f16(false, a0, false, b1, (short)0, acc[1], false, false);
    acc[2] = __builtin_amdgcn_wmma_f32_16x16x32_f16(false, a1, false, b0, (short)0, acc[2], false, false);
    acc[3] = __builtin_amdgcn_wmma_f32_16x16x32_f16(false, a1, false, b1, (short)0, acc[3], false, false);
    __syncthreads();
  }

  // ---- epilogue: n = lane&15, m = vgpr + 8*(lane>>4) ----
  const int ncol = lane & 15;
  const int msub = (lane >> 4) << 3;
#pragma unroll
  for (int sm = 0; sm < 2; ++sm) {
#pragma unroll
    for (int sn = 0; sn < 2; ++sn) {
      const v8f cc = acc[sm * 2 + sn];
      int ng2 = n0 + (wn << 5) + (sn << 4) + ncol;
      if (ng2 >= Ntot) continue;
      int batch2 = ng2 >> p.howo_sh;
      int r2 = ng2 & (HoWo - 1);
      int oh2 = r2 >> p.wo_sh, ow2 = r2 & (p.Wo - 1);
      int mb = m0 + (wm << 5) + (sm << 4) + msub;
#pragma unroll
      for (int u = 0; u < 8; ++u) {
        int m = mb + u;
        if (m >= p.Cout) continue;
        float v = cc[u] * p.scale[m] + p.bias[m];
        size_t oi = ((size_t)(batch2 * p.Cout + m) * p.Ho + oh2) * p.Wo + ow2;
        if (p.res) v += (float)p.res[oi];
        if (p.relu) v = fmaxf(v, 0.f);
        p.y[oi] = (_Float16)v;
      }
    }
  }
}

// ---------------------------------------------------------------------------
// Dynamic routing: one block per (batch, spatial position).
// blockDim = IC*OC (<=256). pred layout: (B*IC, OC*16, Hp, Wp).
// ---------------------------------------------------------------------------
struct RouteP {
  const _Float16* pred;
  _Float16*       v;    // (B, OC*16, Hp, Wp)
  float*          ent;  // global entropy accumulator
  int IC, OC, HW;
  float ent_norm;       // 1 / (ln(OC) * IC * HW)
};

__global__ __launch_bounds__(256) void route_k(RouteP p) {
  __shared__ float sb[256];
  __shared__ float smax[16], ssum[16];
  __shared__ float sacc[256];   // [OC][16]
  __shared__ float sent;
  const int tid = threadIdx.x;
  const int i = tid / p.OC, j = tid - i * p.OC;
  const int b = blockIdx.x / p.HW, yx = blockIdx.x - b * p.HW;
  const int CH = p.OC * 16;
  const _Float16* pp = p.pred + ((size_t)(b * p.IC + i) * CH + j * 16) * p.HW + yx;
  float pv[16];
#pragma unroll
  for (int d = 0; d < 16; ++d) pv[d] = (float)pp[(size_t)d * p.HW];

  float bij = 0.f, c = 0.f;
  for (int it = 0; it < 3; ++it) {
    sb[tid] = bij;
    __syncthreads();
    if (j == 0) {
      float mx = -1e30f;
      for (int jj = 0; jj < p.OC; ++jj) mx = fmaxf(mx, sb[i * p.OC + jj]);
      float s = 0.f;
      for (int jj = 0; jj < p.OC; ++jj) s += expf(sb[i * p.OC + jj] - mx);
      smax[i] = mx; ssum[i] = s;
    }
    __syncthreads();
    c = expf(bij - smax[i]) / ssum[i];
    for (int t = tid; t < CH; t += blockDim.x) sacc[t] = 0.f;
    __syncthreads();
#pragma unroll
    for (int d = 0; d < 16; ++d) atomicAdd(&sacc[j * 16 + d], c * pv[d]);
    __syncthreads();
    if (tid < p.OC) {   // squash s_j
      float n2 = 0.f;
#pragma unroll
      for (int d = 0; d < 16; ++d) { float v = sacc[tid * 16 + d]; n2 += v * v; }
      float sc = n2 / (1.f + n2) / (sqrtf(n2) + 1e-8f);
#pragma unroll
      for (int d = 0; d < 16; ++d) sacc[tid * 16 + d] *= sc;
    }
    __syncthreads();
    if (it < 2) {
      float dp = 0.f;
#pragma unroll
      for (int d = 0; d < 16; ++d) dp += pv[d] * sacc[j * 16 + d];
      bij += dp;
    }
    __syncthreads();
  }
  // entropy of final coupling coefficients
  float e = (c > 1e-30f) ? -c * logf(c) : 0.f;
  if (tid == 0) sent = 0.f;
  __syncthreads();
  atomicAdd(&sent, e);
  __syncthreads();
  if (tid == 0) atomicAdd(p.ent, sent * p.ent_norm);
  if (i == 0) {
#pragma unroll
    for (int d = 0; d < 16; ++d)
      p.v[((size_t)b * CH + j * 16 + d) * p.HW + yx] = (_Float16)sacc[j * 16 + d];
  }
}

// ---------------------------------------------------------------------------
// Small helper kernels
// ---------------------------------------------------------------------------
__global__ void zero_f32_k(float* p, int n) {
  int i = blockIdx.x * blockDim.x + threadIdx.x;
  if (i < n) p[i] = 0.f;
}

__global__ void f32_to_f16_k(const float* s, _Float16* d, int n) {
  int i = blockIdx.x * blockDim.x + threadIdx.x;
  if (i < n) d[i] = (_Float16)s[i];
}

// convert weights (Cout, K) f32 -> (Cout, Kpad) f16, zero padded
__global__ void pad_w_k(const float* s, _Float16* d, int Cout, int K, int Kpad) {
  int i = blockIdx.x * blockDim.x + threadIdx.x;
  int tot = Cout * Kpad;
  if (i >= tot) return;
  int c = i / Kpad, k = i - c * Kpad;
  d[i] = (k < K) ? (_Float16)s[(size_t)c * K + k] : (_Float16)0.f;
}

__global__ void bn_prep_k(const float* gamma, const float* beta, const float* mean,
                          const float* var, const float* convb,
                          float* scale, float* bias, int C) {
  int c = blockIdx.x * blockDim.x + threadIdx.x;
  if (c >= C) return;
  float s = 1.f, b = 0.f;
  if (gamma) { s = gamma[c] * rsqrtf(var[c] + 1e-5f); b = beta[c] - mean[c] * s; }
  float cb = convb ? convb[c] : 0.f;
  scale[c] = s;
  bias[c]  = b + s * cb;
}

__global__ __launch_bounds__(256) void ln_reduce_k(const _Float16* x, float* mu, float* rstd, int CHW) {
  __shared__ float ss[256], sq[256];
  int b = blockIdx.x;
  const _Float16* p = x + (size_t)b * CHW;
  float s = 0.f, q = 0.f;
  for (int i = threadIdx.x; i < CHW; i += 256) { float v = (float)p[i]; s += v; q += v * v; }
  ss[threadIdx.x] = s; sq[threadIdx.x] = q;
  __syncthreads();
  for (int st = 128; st > 0; st >>= 1) {
    if (threadIdx.x < st) { ss[threadIdx.x] += ss[threadIdx.x + st]; sq[threadIdx.x] += sq[threadIdx.x + st]; }
    __syncthreads();
  }
  if (threadIdx.x == 0) {
    float m = ss[0] / (float)CHW;
    float v = sq[0] / (float)CHW - m * m;
    mu[b] = m; rstd[b] = rsqrtf(v + 1e-5f);
  }
}

__global__ void ln_apply_k(const _Float16* x, const float* g, const float* be,
                           const float* mu, const float* rstd, _Float16* y, int CHW, int tot) {
  int i = blockIdx.x * blockDim.x + threadIdx.x;
  if (i >= tot) return;
  int b = i / CHW, c = i - b * CHW;
  y[i] = (_Float16)(((float)x[i] - mu[b]) * rstd[b] * g[c] + be[c]);
}

__global__ void add_f16_k(const _Float16* a, const _Float16* b, _Float16* o, int n) {
  int i = blockIdx.x * blockDim.x + threadIdx.x;
  if (i < n) o[i] = (_Float16)((float)a[i] + (float)b[i]);
}

__global__ void head_k(const _Float16* vc, const float* lw, const float* lb,
                       const float* ent, float* out, int B) {
  int i = blockIdx.x * blockDim.x + threadIdx.x;
  if (i < B * 5) {
    int b = i / 5, c = i - b * 5;
    float s = 0.f;
#pragma unroll
    for (int d = 0; d < 16; ++d) s += (float)vc[(size_t)b * 80 + c * 16 + d] * lw[d];
    out[i] = s + lb[0];
  }
  if (i == 0) out[B * 5] = *ent;
}

// ---------------------------------------------------------------------------
// Host orchestration
// ---------------------------------------------------------------------------
extern "C" void kernel_launch(void* const* d_in, const int* in_sizes, int n_in,
                              void* d_out, int out_size, void* d_ws, size_t ws_size,
                              hipStream_t stream) {
  (void)in_sizes; (void)n_in; (void)out_size; (void)ws_size;

  // --- walk JAX pytree leaves (dict keys sorted alphabetically) ---
  int cur_in = 0;
  auto nx = [&]() -> const float* { return (const float*)d_in[cur_in++]; };
  struct BNr { const float *beta, *gamma, *mean, *var; };
  auto rdBN = [&]() { BNr r; r.beta = nx(); r.gamma = nx(); r.mean = nx(); r.var = nx(); return r; };
  struct CCr { const float* b; BNr bn; const float* w; };
  auto rdCC = [&]() { CCr c; c.b = nx(); c.bn = rdBN(); c.w = nx(); return c; };

  BNr bn1 = rdBN();                 // params['bn1']
  CCr cls = rdCC();                 // params['cls']
  const float* conv1w = nx();       // params['conv1w']
  struct Blk { BNr b1, b2; const float *w1, *w2, *scw; BNr scbn; };
  Blk l1[3];
  for (int i = 0; i < 3; ++i) { l1[i].b1 = rdBN(); l1[i].b2 = rdBN(); l1[i].w1 = nx(); l1[i].w2 = nx(); l1[i].scw = nullptr; }
  Blk l2[4];
  l2[0].b1 = rdBN(); l2[0].b2 = rdBN(); l2[0].w1 = nx(); l2[0].w2 = nx(); l2[0].scbn = rdBN(); l2[0].scw = nx();
  for (int i = 1; i < 4; ++i) { l2[i].b1 = rdBN(); l2[i].b2 = rdBN(); l2[i].w1 = nx(); l2[i].w2 = nx(); l2[i].scw = nullptr; }
  const float* lin_b = nx();
  const float* lin_w = nx();
  const float* ln_bp = nx();
  const float* ln_gp = nx();
  const float* pc_b  = nx();
  const float* pc_w  = nx();
  struct RBr { CCr c1, c2, sc; };
  RBr rb[3];
  for (int i = 0; i < 3; ++i) { rb[i].c1 = rdCC(); rb[i].c2 = rdCC(); rb[i].sc = rdCC(); }
  const float* xin = nx();          // 'x'

  // --- workspace bump allocator ---
  size_t off = 0;
  auto alloc = [&](size_t bytes) -> char* {
    char* p = (char*)d_ws + off;
    off += (bytes + 255) & ~(size_t)255;
    return p;
  };
  float* entAcc = (float*)alloc(sizeof(float));
  float* muArr  = (float*)alloc(128 * sizeof(float));
  float* rsArr  = (float*)alloc(128 * sizeof(float));

  auto tof16 = [&](const float* s, int n) -> _Float16* {
    _Float16* d = (_Float16*)alloc((size_t)n * 2);
    f32_to_f16_k<<<(n + 255) / 256, 256, 0, stream>>>(s, d, n);
    return d;
  };
  // weights: convert + pad K dim to multiple of 32
  auto tow = [&](const float* s, int Cout, int K) -> _Float16* {
    int Kpad = (K + 31) & ~31;
    _Float16* d = (_Float16*)alloc((size_t)Cout * Kpad * 2);
    int tot = Cout * Kpad;
    pad_w_k<<<(tot + 255) / 256, 256, 0, stream>>>(s, d, Cout, K, Kpad);
    return d;
  };
  struct SB { float *s, *b; };
  auto bnp = [&](const BNr* bn, const float* convb, int C) -> SB {
    SB r; r.s = (float*)alloc((size_t)C * 4); r.b = (float*)alloc((size_t)C * 4);
    bn_prep_k<<<(C + 63) / 64, 64, 0, stream>>>(bn ? bn->gamma : nullptr, bn ? bn->beta : nullptr,
                                                bn ? bn->mean : nullptr, bn ? bn->var : nullptr,
                                                convb, r.s, r.b, C);
    return r;
  };

  const size_t ACT = 8388608;                 // 128*64*32*32 == 128*256*16*16 elems
  _Float16* bufs[5];
  for (int i = 0; i < 5; ++i) bufs[i] = (_Float16*)alloc(ACT * 2);
  _Float16* pred = (_Float16*)alloc((size_t)2048 * 256 * 64 * 2);

  zero_f32_k<<<1, 32, 0, stream>>>(entAcc, 1);

  auto ilog2 = [](int v) { int s = 0; while ((1 << s) < v) ++s; return s; };

  auto conv = [&](const _Float16* x, const _Float16* w, SB sb, const _Float16* res, _Float16* y,
                  int N, int Cin, int H, int Cout, int kh, int stride, int pad, int relu) {
    ConvP p;
    p.x = x; p.w = w; p.scale = sb.s; p.bias = sb.b; p.res = res; p.y = y;
    p.N = N; p.Cin = Cin; p.H = H; p.W = H; p.Cout = Cout;
    p.kh = kh; p.kw = kh;
    p.K = Cin * kh * kh;
    p.Kpad = (p.K + 31) & ~31;
    p.stride = stride; p.pad = pad;
    p.Ho = (H + 2 * pad - kh) / stride + 1; p.Wo = p.Ho;
    p.relu = relu;
    p.howo_sh = ilog2(p.Ho * p.Wo);
    p.wo_sh   = ilog2(p.Wo);
    int khw = kh * kh;
    p.kinv_khw = (unsigned)((65536 + khw - 1) / khw);
    p.kinv_kw  = (unsigned)((65536 + kh - 1) / kh);
    int Ntot = N * p.Ho * p.Wo;
    dim3 g((unsigned)((Ntot + 127) / 128), (unsigned)((Cout + 63) / 64));
    conv_wmma<<<g, 256, 0, stream>>>(p);
  };

  // ---------------- backbone ----------------
  _Float16* xh = tof16(xin, 128 * 32 * 32);
  _Float16 *cur = bufs[0], *t1 = bufs[1], *t2 = bufs[2], *e1 = bufs[3], *e2 = bufs[4];

  conv(xh, tow(conv1w, 64, 9), bnp(&bn1, nullptr, 64), nullptr, cur, 128, 1, 32, 64, 3, 1, 1, 1);

  for (int i = 0; i < 3; ++i) {   // layer1, identity shortcuts
    conv(cur, tow(l1[i].w1, 64, 64 * 9), bnp(&l1[i].b1, nullptr, 64), nullptr, t1, 128, 64, 32, 64, 3, 1, 1, 1);
    conv(t1,  tow(l1[i].w2, 64, 64 * 9), bnp(&l1[i].b2, nullptr, 64), cur,     t2, 128, 64, 32, 64, 3, 1, 1, 1);
    _Float16* tmp = cur; cur = t2; t2 = tmp;
  }
  // layer2 block0 (stride-2, projected shortcut)
  conv(cur, tow(l2[0].scw, 128, 64),     bnp(&l2[0].scbn, nullptr, 128), nullptr, t1, 128, 64, 32, 128, 1, 2, 0, 0);
  conv(cur, tow(l2[0].w1, 128, 64 * 9),  bnp(&l2[0].b1, nullptr, 128),   nullptr, t2, 128, 64, 32, 128, 3, 2, 1, 1);
  conv(t2,  tow(l2[0].w2, 128, 128 * 9), bnp(&l2[0].b2, nullptr, 128),   t1,      cur, 128, 128, 16, 128, 3, 1, 1, 1);
  for (int i = 1; i < 4; ++i) {
    conv(cur, tow(l2[i].w1, 128, 128 * 9), bnp(&l2[i].b1, nullptr, 128), nullptr, t1, 128, 128, 16, 128, 3, 1, 1, 1);
    conv(t1,  tow(l2[i].w2, 128, 128 * 9), bnp(&l2[i].b2, nullptr, 128), cur,     t2, 128, 128, 16, 128, 3, 1, 1, 1);
    _Float16* tmp = cur; cur = t2; t2 = tmp;
  }
  // primary caps 1x1 128->256 (+bias), then layernorm over (C,H,W)
  conv(cur, tow(pc_w, 256, 128), bnp(nullptr, pc_b, 256), nullptr, t1, 128, 128, 16, 256, 1, 1, 0, 0);
  ln_reduce_k<<<128, 256, 0, stream>>>(t1, muArr, rsArr, 256 * 16 * 16);
  {
    int tot = 128 * 256 * 16 * 16;
    ln_apply_k<<<(tot + 255) / 256, 256, 0, stream>>>(t1, ln_gp, ln_bp, muArr, rsArr, t2, 256 * 16 * 16, tot);
  }

  // ---------------- capsule stages ----------------
  auto caps_conv = [&](const _Float16* in, CCr& cc, int Cout, int H, int kh, int stride, int pad,
                       _Float16* vout, int OC) -> int {
    conv(in, tow(cc.w, Cout, 16 * kh * kh), bnp(&cc.bn, cc.b, Cout), nullptr, pred,
         2048, 16, H, Cout, kh, stride, pad, 0);
    int Ho = (H + 2 * pad - kh) / stride + 1;
    int HW = Ho * Ho;
    RouteP rp;
    rp.pred = pred; rp.v = vout; rp.ent = entAcc;
    rp.IC = 16; rp.OC = OC; rp.HW = HW;
    rp.ent_norm = 1.0f / (logf((float)OC) * 16.0f * (float)HW);
    route_k<<<128 * HW, 16 * OC, 0, stream>>>(rp);
    return Ho;
  };
  auto add16 = [&](const _Float16* a, const _Float16* b, _Float16* o, int n) {
    add_f16_k<<<(n + 255) / 256, 256, 0, stream>>>(a, b, o, n);
  };

  _Float16* vin = t2;          // caps after layernorm, (B,256,16,16)
  _Float16* capsNext = e2;
  int H = 16;
  for (int r = 0; r < 3; ++r) {
    int Hm = caps_conv(vin, rb[r].c1, 256, H, 3, 2, 1, cur, 16);
    caps_conv(cur, rb[r].c2, 256, Hm, 3, 1, 1, t1, 16);
    caps_conv(vin, rb[r].sc, 256, H, 1, 2, 0, e1, 16);
    add16(t1, e1, capsNext, 128 * 256 * Hm * Hm);
    _Float16* tmp = vin; vin = capsNext; capsNext = tmp;
    H = Hm;                    // 16 -> 8 -> 4 -> 2
  }
  // class capsules: 16 caps -> 5 caps, k=2 -> 1x1
  caps_conv(vin, cls, 80, 2, 2, 1, 0, cur, 5);

  // final linear head + entropy
  head_k<<<3, 256, 0, stream>>>(cur, lin_w, lin_b, entAcc, (float*)d_out, 128);
}
